// fb_SHD_Linear_14989435863743
// MI455X (gfx1250) — matrix operand order repaired and checked by reading, not compiled, not run on hardware
//
#include <hip/hip_runtime.h>
#include <hip/hip_bf16.h>

// MI455X / gfx1250, wave32. Three kernels:
//  0) w1t:      one-shot f16 transpose of W1 into workspace (L2-resident B matrix)
//  1) h1_gemm:  H1[t,b,:] = x[b,t,:] @ W1 + (b1 + rcb1) -- barrier-free WMMA K loop
//  2) lif_scan: persistent recurrent scan -- register-pipelined H1, split WMMA chains

typedef __attribute__((ext_vector_type(16))) _Float16 v16h;
typedef __attribute__((ext_vector_type(8)))  _Float16 v8h;
typedef __attribute__((ext_vector_type(8)))  float    v8f;

#define BATCH 256
#define TSTEPS 250
#define DIN 700
#define KPAD 704     // DIN padded to multiple of 32
#define APITCH 712   // LDS pitch for A panel (bank spread)
#define HID 128
#define DOUT 20
#define WPITCH 136   // f16 pitch for transposed weights in LDS (bank spread)

__device__ __forceinline__ v8f wmma_f16(v16h a, v16h b, v8f c) {
  // v_wmma_f32_16x16x32_f16 : D = A(16x32) x B(32x16) + C(16x16 f32)
  return __builtin_amdgcn_wmma_f32_16x16x32_f16(false, a, false, b, (short)0, c,
                                                false, false);
}

__device__ __forceinline__ v16h combine8(v8h lo, v8h hi) {
  v16h r;
#pragma unroll
  for (int i = 0; i < 8; i++) { r[i] = lo[i]; r[8 + i] = hi[i]; }
  return r;
}

// A-matrix 16x32 f16 fragment from row-major LDS tile (rows pitch `pitch` halves).
// lane<16 : M=lane,   K = k0+0..15
// lane>=16: M=lane-16, K = k0+16..31
__device__ __forceinline__ v16h load_a_frag(const _Float16* base, int pitch,
                                            int m, int sel, int k0) {
  const _Float16* p0 = base + m * pitch + k0 + sel * 16;  // 16B aligned
  return combine8(*(const v8h*)p0, *(const v8h*)(p0 + 8));
}

// B-matrix 32x16 f16 fragment from TRANSPOSED weights Wt[n][k] in LDS.
// lane<16 : col c, K = k0+0..15 ; lane>=16 : col c, K = k0+16..31
__device__ __forceinline__ v16h load_b_frag(const _Float16* wt, int pitch,
                                            int c, int sel, int k0) {
  const _Float16* p = wt + c * pitch + k0 + sel * 16;     // 16B aligned
  return combine8(*(const v8h*)p, *(const v8h*)(p + 8));
}

// ---------------- Kernel 0: W1t[n][k] = f16(W1[k][n]), zero-padded K ----------------
__global__ __launch_bounds__(256) void w1t_kernel(const float* __restrict__ W1,
                                                  _Float16* __restrict__ W1t) {
  int idx = blockIdx.x * 256 + threadIdx.x;   // over 128*704
  if (idx >= HID * KPAD) return;
  int n = idx / KPAD, k = idx - n * KPAD;
  W1t[idx] = (k < DIN) ? (_Float16)W1[(size_t)k * HID + n] : (_Float16)0.0f;
}

// ---------------- Kernel 1: H1 = X @ W1 + (b1 + rcb1) ----------------
// Grid: 4000 blocks (one 16-row M tile, M = t*256 + b), 256 threads (8 waves).
// A panel staged once in LDS (padded); B fragments streamed from L2-resident W1t.
__global__ __launch_bounds__(256) void h1_gemm_kernel(
    const float* __restrict__ x, const _Float16* __restrict__ W1t,
    const float* __restrict__ b1, const float* __restrict__ rcb1,
    float* __restrict__ H1) {
  __shared__ _Float16 sA[16 * APITCH];    // x panel [16][712] f16, zero-padded tail

  const int tid  = threadIdx.x;
  const int wave = tid >> 5;
  const int lane = tid & 31;
  const int sel  = lane >> 4;
  const int lc   = lane & 15;
  const int c    = wave * 16 + lc;        // output column 0..127

  const int m0 = blockIdx.x * 16;         // M = t*256 + b  (256 % 16 == 0 -> same t)
  const int t  = m0 >> 8;
  const int b0 = m0 & 255;

  // one-time coalesced staging of the full A panel (fp32 -> f16), pad k>=700 with 0
  for (int idx = tid; idx < 16 * APITCH; idx += 256) {
    int r = idx / APITCH, k = idx - r * APITCH;
    float v = (k < DIN) ? x[((size_t)(b0 + r) * TSTEPS + t) * DIN + k] : 0.0f;
    sA[idx] = (_Float16)v;
  }
  __syncthreads();

  const _Float16* wcol = W1t + (size_t)c * KPAD + sel * 16;  // 32B aligned
  v8f acc = {};
#pragma unroll 2
  for (int k0 = 0; k0 < KPAD; k0 += 32) {
    v16h a = load_a_frag(sA, APITCH, lc, sel, k0);
    v16h b = *(const v16h*)(wcol + k0);   // 2x global_load_b128, L2-resident
    acc = wmma_f16(a, b, acc);
  }

  const float bias = b1[c] + rcb1[c];     // fold b1 + rcb1 into H1
#pragma unroll
  for (int r = 0; r < 8; r++) {
    int row = r + 8 * sel;                // C layout: M = r + 8*(lane>=16), N = lane&15
    H1[((size_t)t * BATCH + b0 + row) * HID + c] = acc[r] + bias;
  }
}

// ---------------- Kernel 2: recurrent LIF scan ----------------
// Grid: 16 blocks (16 batch rows each), 256 threads (8 waves; wave j owns cols 16j..16j+15).
// Recurrent weights f16-transposed in LDS; v1/v2/readout acc in regs in WMMA C-layout.
// H1 C-tiles are software-pipelined through registers: loads for step t+1 issue mid-step-t
// and are only waited on at loop bottom (workgroup barriers on gfx1250 drain DScnt, not LOADcnt).
__global__ __launch_bounds__(256) void lif_scan_kernel(
    const float* __restrict__ rcW1, const float* __restrict__ W2,
    const float* __restrict__ rcW2, const float* __restrict__ W3,
    const float* __restrict__ b2, const float* __restrict__ rcb2,
    const float* __restrict__ b3, const float* __restrict__ H1,
    float* __restrict__ out) {
  __shared__ _Float16 sRC1[HID * WPITCH];   // rcW1^T  [n][k]
  __shared__ _Float16 sW2 [HID * WPITCH];   // W2^T
  __shared__ _Float16 sRC2[HID * WPITCH];   // rcW2^T
  __shared__ _Float16 sW3 [32 * WPITCH];    // W3^T padded to 32 cols
  __shared__ _Float16 sY1 [16 * HID];       // spike vectors, row-major f16
  __shared__ _Float16 sY2 [16 * HID];

  const int tid  = threadIdx.x;
  const int wave = tid >> 5;
  const int lane = tid & 31;
  const int sel  = lane >> 4;
  const int lc   = lane & 15;
  const int c    = wave * 16 + lc;
  const int b0   = blockIdx.x * 16;

  // cooperative f16 transpose of weights into LDS (one-time)
  for (int idx = tid; idx < HID * HID; idx += 256) {
    int k = idx >> 7, n = idx & 127;
    sRC1[n * WPITCH + k] = (_Float16)rcW1[idx];
    sW2 [n * WPITCH + k] = (_Float16)W2[idx];
    sRC2[n * WPITCH + k] = (_Float16)rcW2[idx];
  }
  for (int idx = tid; idx < 32 * HID; idx += 256) {
    int n = idx >> 7, k = idx & 127;
    sW3[n * WPITCH + k] = (n < DOUT) ? (_Float16)W3[(size_t)k * DOUT + n]
                                     : (_Float16)0.0f;
  }
  for (int idx = tid; idx < 16 * HID; idx += 256) {
    sY1[idx] = (_Float16)0.0f;
    sY2[idx] = (_Float16)0.0f;
  }
  __syncthreads();

  const float bias2 = b2[c] + rcb2[c];
  v8f v1   = {};
  v8f v2   = {};
  v8f acc0 = {};                            // readout accumulators (waves 0,1),
  v8f acc1 = {};                            // split to halve the serial chain

  // prologue: load H1(t=0) C-tile into registers
  v8f hc;
#pragma unroll
  for (int r = 0; r < 8; r++)
    hc[r] = H1[((size_t)0 * BATCH + b0 + r + 8 * sel) * HID + c];

  for (int t = 0; t < TSTEPS; t++) {
    // ---- layer 1: u1 = H1(t) + y1_prev @ rcW1  (2 independent WMMA chains) ----
    v8f u1a = hc, u1b = {};
    u1a = wmma_f16(load_a_frag(sY1, HID, lc, sel, 0),
                   load_b_frag(sRC1, WPITCH, c, sel, 0), u1a);
    u1b = wmma_f16(load_a_frag(sY1, HID, lc, sel, 64),
                   load_b_frag(sRC1, WPITCH, c, sel, 64), u1b);
    u1a = wmma_f16(load_a_frag(sY1, HID, lc, sel, 32),
                   load_b_frag(sRC1, WPITCH, c, sel, 32), u1a);
    u1b = wmma_f16(load_a_frag(sY1, HID, lc, sel, 96),
                   load_b_frag(sRC1, WPITCH, c, sel, 96), u1b);

    // issue next step's H1 loads now; waited on only at loop bottom
    int tn = (t + 1 < TSTEPS) ? t + 1 : t;  // clamp (last iter re-loads, unused)
    const float* hp = H1 + ((size_t)tn * BATCH + b0) * HID;
    v8f hn;
#pragma unroll
    for (int r = 0; r < 8; r++) hn[r] = hp[(r + 8 * sel) * HID + c];

    __syncthreads();                        // all reads of y1_prev done

    // LIF 1: v = (v+u)/2 ; spike ; hard reset
#pragma unroll
    for (int r = 0; r < 8; r++) {
      float v = 0.5f * (v1[r] + (u1a[r] + u1b[r]));
      float s = (v >= 1.0f) ? 1.0f : 0.0f;
      v1[r] = v * (1.0f - s);
      sY1[(r + 8 * sel) * HID + c] = (_Float16)s;
    }
    __syncthreads();                        // y1_new visible

    // ---- layer 2: u2 = (b2+rcb2) + y1_new @ W2 + y2_prev @ rcW2 ----
    // 4 independent WMMA chains of depth 2
    v8f uA, uB = {}, uC = {}, uD = {};
#pragma unroll
    for (int r = 0; r < 8; r++) uA[r] = bias2;
#pragma unroll
    for (int k = 0; k < 2; k++) {
      uA = wmma_f16(load_a_frag(sY1, HID, lc, sel, k * 32),
                    load_b_frag(sW2, WPITCH, c, sel, k * 32), uA);
      uC = wmma_f16(load_a_frag(sY2, HID, lc, sel, k * 32),
                    load_b_frag(sRC2, WPITCH, c, sel, k * 32), uC);
      uB = wmma_f16(load_a_frag(sY1, HID, lc, sel, 64 + k * 32),
                    load_b_frag(sW2, WPITCH, c, sel, 64 + k * 32), uB);
      uD = wmma_f16(load_a_frag(sY2, HID, lc, sel, 64 + k * 32),
                    load_b_frag(sRC2, WPITCH, c, sel, 64 + k * 32), uD);
    }
    __syncthreads();                        // all reads of y2_prev done

    // LIF 2
#pragma unroll
    for (int r = 0; r < 8; r++) {
      float v = 0.5f * (v2[r] + ((uA[r] + uB[r]) + (uC[r] + uD[r])));
      float s = (v >= 1.0f) ? 1.0f : 0.0f;
      v2[r] = v * (1.0f - s);
      sY2[(r + 8 * sel) * HID + c] = (_Float16)s;
    }
    __syncthreads();                        // y2_new visible

    // ---- readout: acc += y2_new @ W3 (waves 0,1 only; wave-uniform branch) ----
    if (wave < 2) {
      acc0 = wmma_f16(load_a_frag(sY2, HID, lc, sel, 0),
                      load_b_frag(sW3, WPITCH, c, sel, 0), acc0);
      acc1 = wmma_f16(load_a_frag(sY2, HID, lc, sel, 64),
                      load_b_frag(sW3, WPITCH, c, sel, 64), acc1);
      acc0 = wmma_f16(load_a_frag(sY2, HID, lc, sel, 32),
                      load_b_frag(sW3, WPITCH, c, sel, 32), acc0);
      acc1 = wmma_f16(load_a_frag(sY2, HID, lc, sel, 96),
                      load_b_frag(sW3, WPITCH, c, sel, 96), acc1);
    }

    hc = hn;                                // s_wait_loadcnt lands here, fully hidden
  }

  if (wave < 2 && c < DOUT) {
    float bb = b3[c] * (float)TSTEPS;       // out_t added b3 every step
#pragma unroll
    for (int r = 0; r < 8; r++)
      out[(size_t)(b0 + r + 8 * sel) * DOUT + c] = (acc0[r] + acc1[r]) + bb;
  }
}

extern "C" void kernel_launch(void* const* d_in, const int* in_sizes, int n_in,
                              void* d_out, int out_size, void* d_ws, size_t ws_size,
                              hipStream_t stream) {
  // setup_inputs order: x, W1, b1, rcW1, rcb1, W2, b2, rcW2, rcb2, W3, b3
  const float* x    = (const float*)d_in[0];
  const float* W1   = (const float*)d_in[1];
  const float* b1   = (const float*)d_in[2];
  const float* rcW1 = (const float*)d_in[3];
  const float* rcb1 = (const float*)d_in[4];
  const float* W2   = (const float*)d_in[5];
  const float* b2   = (const float*)d_in[6];
  const float* rcW2 = (const float*)d_in[7];
  const float* rcb2 = (const float*)d_in[8];
  const float* W3   = (const float*)d_in[9];
  const float* b3   = (const float*)d_in[10];
  float* out = (float*)d_out;

  // workspace layout: H1 fp32 [T][B][H] (32.77 MB), then W1t f16 [128][704] (176 KB)
  float*    H1  = (float*)d_ws;
  _Float16* W1t = (_Float16*)((char*)d_ws + (size_t)TSTEPS * BATCH * HID * sizeof(float));

  // Phase 0: one-shot f16 transpose of W1 (stays L2-resident for phase 1)
  w1t_kernel<<<dim3((HID * KPAD + 255) / 256), dim3(256), 0, stream>>>(W1, W1t);

  // Phase 1: big parallel GEMM (memory-bound; f16 WMMA, barrier-free K loop)
  h1_gemm_kernel<<<dim3((TSTEPS * BATCH) / 16), dim3(256), 0, stream>>>(
      x, W1t, b1, rcb1, H1);

  // Phase 2: sequential scan, 16 persistent batch-slice workgroups
  lif_scan_kernel<<<dim3(BATCH / 16), dim3(256), 0, stream>>>(
      rcW1, W2, rcW2, W3, b2, rcb2, b3, H1, out);
}
